// NetworkCBF_86131274154568
// MI455X (gfx1250) — compile-verified, matrix operand order checked by reading, not run on hardware
//
#include <hip/hip_runtime.h>

#define NAGENTS 4096
#define TOPK 12
#define NK (NAGENTS * TOPK) // 49152

typedef __attribute__((ext_vector_type(16))) _Float16 v16h;
typedef __attribute__((ext_vector_type(8)))  _Float16 v8h;
typedef __attribute__((ext_vector_type(8)))  float    v8f;
typedef __attribute__((ext_vector_type(4)))  float    v4f;

// ---------------------------------------------------------------------------
// Kernel 1: per-row top-12 nearest neighbors.
// Key = (bits(d2) << 32) | j  -> min-key == smallest distance, lowest index
// tie-break, matching jax.lax.top_k(-dn).
// ---------------------------------------------------------------------------
__global__ void __launch_bounds__(256) topk_kernel(const float* __restrict__ x,
                                                   int* __restrict__ idxOut) {
  const int row = blockIdx.x;
  const int t = threadIdx.x;
  const float* xr = x + (size_t)row * NAGENTS * 4;

  unsigned long long key[16];
#pragma unroll
  for (int i = 0; i < 16; ++i) {
    int j = i * 256 + t;
    v4f f = *(const v4f*)(xr + (size_t)j * 4); // b128 load, only xy used
    float d2 = f[0] * f[0] + f[1] * f[1] + 2.0e-6f;
    key[i] = ((unsigned long long)__float_as_uint(d2) << 32) | (unsigned)j;
  }

  __shared__ unsigned long long red[256];
  for (int round = 0; round < TOPK; ++round) {
    unsigned long long m = ~0ull;
#pragma unroll
    for (int i = 0; i < 16; ++i) m = key[i] < m ? key[i] : m;
    red[t] = m;
    __syncthreads();
    for (int s = 128; s > 0; s >>= 1) {
      if (t < s) {
        unsigned long long o = red[t + s];
        if (o < red[t]) red[t] = o;
      }
      __syncthreads();
    }
    unsigned long long win = red[0];
    __syncthreads(); // red[] reused next round
    if (t == 0) idxOut[row * TOPK + round] = (int)(unsigned)win;
#pragma unroll
    for (int i = 0; i < 16; ++i)
      if (key[i] == win) key[i] = ~0ull;
  }
}

// ---------------------------------------------------------------------------
// Fragment load for v_wmma_f32_16x16x32_f16 (16-bit A/B layout):
//   lane < 16 : K = {0..7} (h0..7),  {16..23} (h8..15)
//   lane >=16 : K = {8..15},         {24..31}
// rowp points at the start of the matrix row (A: row M=lane&15; B: col N).
// ---------------------------------------------------------------------------
__device__ __forceinline__ v16h frag_ld(const _Float16* rowp, int kchunk, int hi) {
  const _Float16* p = rowp + kchunk * 32 + (hi ? 8 : 0);
  v8h lo = *(const v8h*)p;        // ds_load_b128
  v8h hh = *(const v8h*)(p + 16); // ds_load_b128
  v16h r;
#pragma unroll
  for (int i = 0; i < 8; ++i) { r[i] = lo[i]; r[i + 8] = hh[i]; }
  return r;
}

#define WMMA_F16(a, b, c) \
  __builtin_amdgcn_wmma_f32_16x16x32_f16(false, (a), false, (b), (short)0, (c), false, false)

// ---------------------------------------------------------------------------
// Kernel 2: gather top-k features + 4-layer MLP via WMMA.
// 4 waves per block; each wave owns one 16-row tile of the 49152 (n,k) rows.
// ---------------------------------------------------------------------------
__global__ void __launch_bounds__(128) mlp_kernel(
    const float* __restrict__ x, const int* __restrict__ idx,
    const float* __restrict__ rs,
    const float* __restrict__ W1, const float* __restrict__ b1,
    const float* __restrict__ W2, const float* __restrict__ b2,
    const float* __restrict__ W3, const float* __restrict__ b3,
    const float* __restrict__ W4, const float* __restrict__ b4,
    float* __restrict__ dout) {
  __shared__ __align__(16) _Float16 W1h[64 * 32];   // [out=64][in pad 32]
  __shared__ __align__(16) _Float16 W2h[128 * 64];  // [out=128][in=64]
  __shared__ __align__(16) _Float16 W3h[64 * 128];  // [out=64][in=128]
  __shared__ __align__(16) _Float16 W4h[16 * 64];   // row0 = W4, rows 1..15 = 0
  __shared__ float bias1[64], bias2[128], bias3[64];
  __shared__ float bias4;
  __shared__ __align__(16) _Float16 act[4][16 * 128]; // per-wave activation tile
  __shared__ float maskb[4][16];

  const int tid = threadIdx.x;

  // Cooperative f32 -> f16 weight staging (once per block).
  for (int i = tid; i < 64 * 32; i += 128) {
    int o = i >> 5, c = i & 31;
    W1h[i] = (_Float16)(c < 6 ? W1[o * 6 + c] : 0.0f);
  }
  for (int i = tid; i < 128 * 64; i += 128) W2h[i] = (_Float16)W2[i];
  for (int i = tid; i < 64 * 128; i += 128) W3h[i] = (_Float16)W3[i];
  for (int i = tid; i < 16 * 64; i += 128) W4h[i] = (_Float16)((i < 64) ? W4[i] : 0.0f);
  if (tid < 64) { bias1[tid] = b1[tid]; bias3[tid] = b3[tid]; }
  if (tid < 128) bias2[tid] = b2[tid];
  if (tid == 0) bias4 = b4[0];
  __syncthreads();

  const int w = tid >> 5;
  const int lane = tid & 31;
  const int nlo = lane & 15;
  const int hi = lane >> 4;
  const int tile = blockIdx.x * 4 + w;
  _Float16* A = act[w];
  float* mrow = maskb[w];

  float* outp = dout;
  float* maskp = dout + NK;
  int* indp = (int*)(dout + 2 * NK);

  // ---- Gather: lanes 0..15 each build one feature row [6 feats, pad 32] ----
  if (lane < 16) {
    int row = tile * 16 + lane;
    int n = row / TOPK;
    int j = idx[row];
    const float* xp = x + ((size_t)n * NAGENTS + j) * 4;
    v4f f = *(const v4f*)xp;
    float dist = sqrtf(f[0] * f[0] + f[1] * f[1] + 2.0e-4f);
    float mk = (dist <= 1.0f) ? 1.0f : 0.0f;
    _Float16* rp = A + lane * 32;
    rp[0] = (_Float16)f[0];
    rp[1] = (_Float16)f[1];
    rp[2] = (_Float16)f[2];
    rp[3] = (_Float16)f[3];
    rp[4] = (_Float16)((n == j) ? 1.0f : 0.0f);
    rp[5] = (_Float16)(dist - rs[0]);
#pragma unroll
    for (int c = 6; c < 32; ++c) rp[c] = (_Float16)0.0f;
    mrow[lane] = mk;
    maskp[row] = mk;                 // output 2: mask
    indp[2 * row + 0] = n;           // output 3: indices
    indp[2 * row + 1] = j;
  }
  asm volatile("s_wait_dscnt 0" ::: "memory"); // gather stores -> frag loads

  // ---- Layer 1: [16x32(6)] x W1^T -> [16x64], bias+ReLU, restage stride 64
  {
    v16h a = frag_ld(A + nlo * 32, 0, hi);
#pragma unroll
    for (int nc = 0; nc < 4; ++nc) {
      v16h b = frag_ld(W1h + (nc * 16 + nlo) * 32, 0, hi);
      v8f c = {};
      c = WMMA_F16(a, b, c);
      float bb = bias1[nc * 16 + nlo];
      int mb = hi * 8;
#pragma unroll
      for (int v = 0; v < 8; ++v) {
        float y = c[v] + bb;
        y = y > 0.0f ? y : 0.0f;
        A[(mb + v) * 64 + nc * 16 + nlo] = (_Float16)y;
      }
    }
  }

  // ---- Layer 2: [16x64] x W2^T -> [16x128], restage stride 128 ----
  {
    v16h a0 = frag_ld(A + nlo * 64, 0, hi);
    v16h a1 = frag_ld(A + nlo * 64, 1, hi);
#pragma unroll
    for (int nc = 0; nc < 8; ++nc) {
      const _Float16* wr = W2h + (nc * 16 + nlo) * 64;
      v8f c = {};
      c = WMMA_F16(a0, frag_ld(wr, 0, hi), c);
      c = WMMA_F16(a1, frag_ld(wr, 1, hi), c);
      float bb = bias2[nc * 16 + nlo];
      int mb = hi * 8;
#pragma unroll
      for (int v = 0; v < 8; ++v) {
        float y = c[v] + bb;
        y = y > 0.0f ? y : 0.0f;
        A[(mb + v) * 128 + nc * 16 + nlo] = (_Float16)y;
      }
    }
  }

  // ---- Layer 3: [16x128] x W3^T -> [16x64], restage stride 64 ----
  {
    v16h a0 = frag_ld(A + nlo * 128, 0, hi);
    v16h a1 = frag_ld(A + nlo * 128, 1, hi);
    v16h a2 = frag_ld(A + nlo * 128, 2, hi);
    v16h a3 = frag_ld(A + nlo * 128, 3, hi);
#pragma unroll
    for (int nc = 0; nc < 4; ++nc) {
      const _Float16* wr = W3h + (nc * 16 + nlo) * 128;
      v8f c = {};
      c = WMMA_F16(a0, frag_ld(wr, 0, hi), c);
      c = WMMA_F16(a1, frag_ld(wr, 1, hi), c);
      c = WMMA_F16(a2, frag_ld(wr, 2, hi), c);
      c = WMMA_F16(a3, frag_ld(wr, 3, hi), c);
      float bb = bias3[nc * 16 + nlo];
      int mb = hi * 8;
#pragma unroll
      for (int v = 0; v < 8; ++v) {
        float y = c[v] + bb;
        y = y > 0.0f ? y : 0.0f;
        A[(mb + v) * 64 + nc * 16 + nlo] = (_Float16)y;
      }
    }
  }

  // ---- Layer 4: [16x64] x W4^T (col 0 only) -> out * mask ----
  {
    v16h a0 = frag_ld(A + nlo * 64, 0, hi);
    v16h a1 = frag_ld(A + nlo * 64, 1, hi);
    const _Float16* wr = W4h + nlo * 64;
    v8f c = {};
    c = WMMA_F16(a0, frag_ld(wr, 0, hi), c);
    c = WMMA_F16(a1, frag_ld(wr, 1, hi), c);
    if (nlo == 0) { // lanes 0 (rows 0..7) and 16 (rows 8..15) hold column N=0
      int mb = hi * 8;
#pragma unroll
      for (int v = 0; v < 8; ++v) {
        int m = mb + v;
        outp[tile * 16 + m] = (c[v] + bias4) * mrow[m];
      }
    }
  }
}

// ---------------------------------------------------------------------------
extern "C" void kernel_launch(void* const* d_in, const int* in_sizes, int n_in,
                              void* d_out, int out_size, void* d_ws, size_t ws_size,
                              hipStream_t stream) {
  const float* x  = (const float*)d_in[0];
  const float* r  = (const float*)d_in[1];
  const float* W1 = (const float*)d_in[2];
  const float* b1 = (const float*)d_in[3];
  const float* W2 = (const float*)d_in[4];
  const float* b2 = (const float*)d_in[5];
  const float* W3 = (const float*)d_in[6];
  const float* b3 = (const float*)d_in[7];
  const float* W4 = (const float*)d_in[8];
  const float* b4 = (const float*)d_in[9];

  int* idxbuf = (int*)d_ws; // NK int32 indices

  topk_kernel<<<NAGENTS, 256, 0, stream>>>(x, idxbuf);
  mlp_kernel<<<NK / 16 / 4, 128, 0, stream>>>(x, idxbuf, r, W1, b1, W2, b2,
                                              W3, b3, W4, b4, (float*)d_out);
}
